// Graph_25460566131066
// MI455X (gfx1250) — matrix operand order repaired and checked by reading, not compiled, not run on hardware
//
#include <hip/hip_runtime.h>
#include <hip/hip_bf16.h>
#include <math.h>

#define N_NODES    50000
#define N_PAD      50048              // round up to multiple of 64 (GEMM block rows)
#define N_EDGES    640000
#define IN_DIM     5
#define HID        128
#define NUM_GRAPHS 512
#define KDEPTH     (2 * HID)          // 256 (concat of agg|h)
#define LDS_STRIDE (KDEPTH + 8)       // 264 ushorts: +8 pad kills LDS bank conflicts
#define MTILE      64                 // rows per GEMM block (4 row-tiles x 16)

typedef __attribute__((ext_vector_type(16))) __bf16 v16bf;
typedef __attribute__((ext_vector_type(8)))  float  v8f;

__device__ __forceinline__ unsigned short f32_to_bf16_rne(float f) {
  union { float f; unsigned int u; } in; in.f = f;
  unsigned int b = in.u;
  b += 0x7FFFu + ((b >> 16) & 1u);     // round-to-nearest-even
  return (unsigned short)(b >> 16);
}
__device__ __forceinline__ float bf16_to_f32(unsigned int u16) {
  union { unsigned int i; float f; } o; o.i = u16 << 16; return o.f;
}

// ---- weight prep: WcatT[n][k] = transposed [W_rel ; W_root], bf16 -----------
__global__ void concat_weights_bf16_T(const float* __restrict__ Wrel,
                                      const float* __restrict__ Wroot,
                                      unsigned short* __restrict__ WcatT) {
  int i = blockIdx.x * blockDim.x + threadIdx.x;
  if (i >= KDEPTH * HID) return;
  int k = i / HID, f = i % HID;
  float v = (k < HID) ? Wrel[k * HID + f] : Wroot[(k - HID) * HID + f];
  WcatT[f * KDEPTH + k] = f32_to_bf16_rne(v);
}

// ---- edge aggregation, 5-dim (layer 0, f32 x) -------------------------------
__global__ void agg_edges_dim5(const float* __restrict__ x,
                               const int* __restrict__ src,
                               const int* __restrict__ dst,
                               float* __restrict__ agg) {
  int e = blockIdx.x * blockDim.x + threadIdx.x;
  if (e >= N_EDGES) return;
  int s = src[e], d = dst[e];
  #pragma unroll
  for (int k = 0; k < IN_DIM; ++k)
    atomicAdd(&agg[d * IN_DIM + k], x[s * IN_DIM + k]);
}

// ---- layer 0: tiny GEMM (K=5), VALU; writes bf16 activations ----------------
__global__ void layer0(const float* __restrict__ x, const float* __restrict__ agg,
                       const float* __restrict__ Wrel, const float* __restrict__ brel,
                       const float* __restrict__ Wroot, unsigned short* __restrict__ h) {
  int i = blockIdx.x * blockDim.x + threadIdx.x;
  if (i >= N_NODES * HID) return;
  int n = i / HID, f = i % HID;
  float acc = brel[f];
  #pragma unroll
  for (int k = 0; k < IN_DIM; ++k)
    acc = fmaf(agg[n * IN_DIM + k], Wrel[k * HID + f],
          fmaf(x[n * IN_DIM + k], Wroot[k * HID + f], acc));
  h[i] = f32_to_bf16_rne(fmaxf(acc, 0.0f));
}

// ---- edge aggregation, 128-dim: thread = (edge, 4 feats); bf16 gather -------
__global__ void agg_edges_dim128(const unsigned short* __restrict__ h,
                                 const int* __restrict__ src,
                                 const int* __restrict__ dst,
                                 float* __restrict__ agg) {
  long long gid = (long long)blockIdx.x * blockDim.x + threadIdx.x;
  if (gid >= (long long)N_EDGES * 32) return;
  int e  = (int)(gid >> 5);
  int f4 = ((int)gid & 31) * 4;
  int s = src[e], d = dst[e];
  uint2 r = *(const uint2*)(h + (long long)s * HID + f4);   // 4 bf16
  float* ap = agg + (long long)d * HID + f4;
  atomicAdd(ap + 0, bf16_to_f32(r.x & 0xFFFFu));
  atomicAdd(ap + 1, bf16_to_f32(r.x >> 16));
  atomicAdd(ap + 2, bf16_to_f32(r.y & 0xFFFFu));
  atomicAdd(ap + 3, bf16_to_f32(r.y >> 16));
}

// ---- f32 -> bf16 vectorized convert (agg buffer before GEMM) ----------------
__global__ void cvt_f32_bf16_vec4(const float* __restrict__ in,
                                  unsigned short* __restrict__ out, int n4) {
  int i = blockIdx.x * blockDim.x + threadIdx.x;
  if (i >= n4) return;
  float4 v = ((const float4*)in)[i];
  uint2 p;
  p.x = (unsigned)f32_to_bf16_rne(v.x) | ((unsigned)f32_to_bf16_rne(v.y) << 16);
  p.y = (unsigned)f32_to_bf16_rne(v.z) | ((unsigned)f32_to_bf16_rne(v.w) << 16);
  ((uint2*)out)[i] = p;
}

// ---- fused dual-GEMM: out = agg@W_rel + hprev@W_root + b, optional ReLU -----
// A = [agg | hprev] (N_PAD x 256 bf16); WcatT (128 x 256 bf16) staged in padded
// LDS. Block = 8 waves -> 64x128 output (4 row-tiles x 2 col-halves); each wave
// owns a 16x64 strip: per K-chunk 2 global b128 (A) feed 4 v_wmma (B from LDS).
__global__ __launch_bounds__(256) void gconv_wmma(
    const unsigned short* __restrict__ aggb, const unsigned short* __restrict__ hprevb,
    const unsigned short* __restrict__ WcatT, const float* __restrict__ bias,
    unsigned short* __restrict__ outb, int do_relu) {
  extern __shared__ unsigned short sWT[];            // 128 * 264 ushorts
  { // cooperative stage, uint4 = 8 bf16; 128 rows x 32 segments
    const uint4* g = (const uint4*)WcatT;
    for (int idx = threadIdx.x; idx < HID * (KDEPTH / 8); idx += blockDim.x) {
      int row = idx >> 5;                            // /32
      int seg = idx & 31;
      *(uint4*)(sWT + row * LDS_STRIDE + seg * 8) = g[row * (KDEPTH / 8) + seg];
    }
  }
  __syncthreads();

  const int lane  = threadIdx.x & 31;
  const int wave  = threadIdx.x >> 5;                // 0..7
  const int rowt  = wave >> 1;                       // 0..3 row-tile in block
  const int nhalf = wave & 1;                        // 0..1 column half
  const int half  = lane >> 4;                       // ISA lane-half striping
  const int r16   = lane & 15;
  const int m0    = blockIdx.x * MTILE + rowt * 16;
  const int nbase = nhalf * 64;

  __builtin_prefetch((const void*)(hprevb + (long long)(m0 + r16) * HID), 0, 0);

  v8f c[4] = {};
  #pragma unroll
  for (int kk = 0; kk < KDEPTH; kk += 32) {
    const unsigned short* Asrc = (kk < HID) ? aggb : hprevb;
    const int kbase            = (kk < HID) ? kk   : (kk - HID);
    // A fragment (16-bit A map): lane half h, row M = r16,
    // K = kbase + h*8 + {0..7} and kbase + 16 + h*8 + {0..7}  -> two b128 loads
    const unsigned short* arow = Asrc + (long long)(m0 + r16) * HID + kbase + half * 8;
    union { uint4 q[2]; v16bf v; } A;
    A.q[0] = *(const uint4*)(arow);
    A.q[1] = *(const uint4*)(arow + 16);
    // 4 N-tiles share this A fragment -> 4 WMMAs per 2 A loads
    #pragma unroll
    for (int t = 0; t < 4; ++t) {
      const int n = nbase + t * 16 + r16;
      const unsigned short* bp = sWT + n * LDS_STRIDE + kk + half * 8;
      union { uint4 q[2]; v16bf v; } B;
      B.q[0] = *(const uint4*)(bp);
      B.q[1] = *(const uint4*)(bp + 16);
      c[t] = __builtin_amdgcn_wmma_f32_16x16x32_bf16(
          /*neg_a=*/false, A.v, /*neg_b=*/false, B.v,
          /*c_mod=*/(short)0, c[t], /*reuse_a=*/false, /*reuse_b=*/false);
    }
  }

  // C/D map: VGPR j -> M = m0 + half*8 + j, N = nbase + t*16 + r16
  #pragma unroll
  for (int t = 0; t < 4; ++t) {
    const int n  = nbase + t * 16 + r16;
    const float bn = bias[n];
    #pragma unroll
    for (int j = 0; j < 8; ++j) {
      int m = m0 + half * 8 + j;
      float v = c[t][j] + bn;
      if (do_relu) v = fmaxf(v, 0.0f);
      outb[(long long)m * HID + n] = f32_to_bf16_rne(v);
    }
  }
}

// ---- pooling + head ---------------------------------------------------------
__global__ void pool_accum(const unsigned short* __restrict__ h,
                           const int* __restrict__ batch, float* __restrict__ sums) {
  long long i = (long long)blockIdx.x * blockDim.x + threadIdx.x;
  if (i >= (long long)N_NODES * HID) return;
  int nn = (int)(i >> 7), f = (int)(i & 127);
  atomicAdd(&sums[batch[nn] * HID + f], bf16_to_f32(h[i]));
}

__global__ void pool_counts(const int* __restrict__ batch, float* __restrict__ cnt) {
  int nn = blockIdx.x * blockDim.x + threadIdx.x;
  if (nn >= N_NODES) return;
  atomicAdd(&cnt[batch[nn]], 1.0f);
}

__global__ void head(const float* __restrict__ sums, const float* __restrict__ cnt,
                     const float* __restrict__ Wlin, const float* __restrict__ blin,
                     float* __restrict__ out) {
  int g = blockIdx.x * blockDim.x + threadIdx.x;
  if (g >= NUM_GRAPHS) return;
  float inv = 1.0f / fmaxf(cnt[g], 1.0f);
  float acc = blin[0];
  for (int f = 0; f < HID; ++f) acc = fmaf(sums[g * HID + f] * inv, Wlin[f], acc);
  out[g] = 1.0f / (1.0f + expf(-acc));
}

extern "C" void kernel_launch(void* const* d_in, const int* in_sizes, int n_in,
                              void* d_out, int out_size, void* d_ws, size_t ws_size,
                              hipStream_t stream) {
  const float* x      = (const float*)d_in[0];
  const int*   eidx   = (const int*)d_in[1];
  const int*   batch  = (const int*)d_in[2];
  const float* Wrel0  = (const float*)d_in[3];
  const float* brel0  = (const float*)d_in[4];
  const float* Wroot0 = (const float*)d_in[5];
  const float* Wrel1  = (const float*)d_in[6];
  const float* brel1  = (const float*)d_in[7];
  const float* Wroot1 = (const float*)d_in[8];
  const float* Wrel2  = (const float*)d_in[9];
  const float* brel2  = (const float*)d_in[10];
  const float* Wroot2 = (const float*)d_in[11];
  const float* Wlin   = (const float*)d_in[12];
  const float* blin   = (const float*)d_in[13];
  const int* src = eidx;
  const int* dst = eidx + N_EDGES;
  float* out = (float*)d_out;

  // ---- carve workspace (~66 MB needed) ----
  char* ws = (char*)d_ws;
  size_t off = 0;
  auto carve = [&](size_t bytes) -> void* {
    void* p = ws + off;
    off = (off + bytes + 255) & ~(size_t)255;
    return p;
  };
  unsigned short* WcatT1 = (unsigned short*)carve((size_t)KDEPTH * HID * 2);
  unsigned short* WcatT2 = (unsigned short*)carve((size_t)KDEPTH * HID * 2);
  float*          agg0   = (float*)carve((size_t)N_NODES * IN_DIM * 4);
  unsigned short* hA16   = (unsigned short*)carve((size_t)N_PAD * HID * 2);
  unsigned short* hB16   = (unsigned short*)carve((size_t)N_PAD * HID * 2);
  float*          aggF   = (float*)carve((size_t)N_NODES * HID * 4);
  unsigned short* agg16  = (unsigned short*)carve((size_t)N_PAD * HID * 2);
  float*          sums   = (float*)carve((size_t)NUM_GRAPHS * HID * 4);
  float*          cnt    = (float*)carve((size_t)NUM_GRAPHS * 4);
  (void)ws_size; (void)n_in; (void)in_sizes; (void)out_size;

  // zero the 48 pad rows of each bf16 buffer (GEMM reads them; deterministic)
  const size_t padOff   = (size_t)N_NODES * HID;
  const size_t padBytes = (size_t)(N_PAD - N_NODES) * HID * 2;
  hipMemsetAsync(hA16 + padOff, 0, padBytes, stream);
  hipMemsetAsync(hB16 + padOff, 0, padBytes, stream);
  hipMemsetAsync(agg16 + padOff, 0, padBytes, stream);

  // weights -> transposed bf16 concat
  concat_weights_bf16_T<<<(KDEPTH * HID + 255) / 256, 256, 0, stream>>>(Wrel1, Wroot1, WcatT1);
  concat_weights_bf16_T<<<(KDEPTH * HID + 255) / 256, 256, 0, stream>>>(Wrel2, Wroot2, WcatT2);

  // layer 0
  hipMemsetAsync(agg0, 0, (size_t)N_NODES * IN_DIM * 4, stream);
  agg_edges_dim5<<<(N_EDGES + 255) / 256, 256, 0, stream>>>(x, src, dst, agg0);
  layer0<<<(N_NODES * HID + 255) / 256, 256, 0, stream>>>(x, agg0, Wrel0, brel0, Wroot0, hA16);

  const int aggGrid  = (int)(((long long)N_EDGES * 32 + 255) / 256);  // 80000
  const int cvtGrid  = (N_NODES * HID / 4 + 255) / 256;               // 6250
  const int gemmGrid = N_PAD / MTILE;                                  // 782
  const size_t ldsW  = (size_t)HID * LDS_STRIDE * 2;                   // 67584 B

  // layer 1 (ReLU)
  hipMemsetAsync(aggF, 0, (size_t)N_NODES * HID * 4, stream);
  agg_edges_dim128<<<aggGrid, 256, 0, stream>>>(hA16, src, dst, aggF);
  cvt_f32_bf16_vec4<<<cvtGrid, 256, 0, stream>>>(aggF, agg16, N_NODES * HID / 4);
  gconv_wmma<<<gemmGrid, 256, ldsW, stream>>>(agg16, hA16, WcatT1, brel1, hB16, 1);

  // layer 2 (no activation), output overwrites hA16
  hipMemsetAsync(aggF, 0, (size_t)N_NODES * HID * 4, stream);
  agg_edges_dim128<<<aggGrid, 256, 0, stream>>>(hB16, src, dst, aggF);
  cvt_f32_bf16_vec4<<<cvtGrid, 256, 0, stream>>>(aggF, agg16, N_NODES * HID / 4);
  gconv_wmma<<<gemmGrid, 256, ldsW, stream>>>(agg16, hB16, WcatT2, brel2, hA16, 0);

  // mean pool + head
  hipMemsetAsync(sums, 0, (size_t)NUM_GRAPHS * HID * 4, stream);
  hipMemsetAsync(cnt,  0, (size_t)NUM_GRAPHS * 4, stream);
  pool_accum<<<(N_NODES * HID + 255) / 256, 256, 0, stream>>>(hA16, batch, sums);
  pool_counts<<<(N_NODES + 255) / 256, 256, 0, stream>>>(batch, cnt);
  head<<<(NUM_GRAPHS + 255) / 256, 256, 0, stream>>>(sums, cnt, Wlin, blin, out);
}
